// MultiHeadAttention_60559038873660
// MI455X (gfx1250) — matrix operand order repaired and checked by reading, not compiled, not run on hardware
//
#include <hip/hip_runtime.h>

typedef __bf16 bf16_t;
typedef __attribute__((ext_vector_type(16))) __bf16 v16bf;
typedef __attribute__((ext_vector_type(8)))  __bf16 v8bf;
typedef __attribute__((ext_vector_type(8)))  float  v8f;
typedef __attribute__((ext_vector_type(4)))  int    v4i;

#define DIMC 1024
#define NH   16
#define HD   64
#define BB   2
#define LL   2048
#define MMK  2048

__device__ __forceinline__ v8f wmma_bf16(v16bf a, v16bf b, v8f c) {
  // (neg_a, A, neg_b, B, c_mod, C, reuse_a, reuse_b)
  return __builtin_amdgcn_wmma_f32_16x16x32_bf16(false, a, false, b, (short)0, c, false, false);
}
__device__ __forceinline__ v16bf cat8(v8bf a, v8bf b) {
  return __builtin_shufflevector(a, b, 0,1,2,3,4,5,6,7,8,9,10,11,12,13,14,15);
}

// ---- CDNA5 async global->LDS copy (ASYNCcnt-tracked), with safe fallback ----
#if __has_builtin(__builtin_amdgcn_global_load_async_to_lds_b128)
#define HAVE_ASYNC_LDS 1
typedef __attribute__((address_space(1))) v4i g_v4i;   // global int4
typedef __attribute__((address_space(3))) v4i l_v4i;   // LDS int4
#else
#define HAVE_ASYNC_LDS 0
#endif

__device__ __forceinline__ void async_copy16(void* lds_dst, const void* gsrc) {
#if HAVE_ASYNC_LDS
  __builtin_amdgcn_global_load_async_to_lds_b128((g_v4i*)gsrc, (l_v4i*)lds_dst, 0, 0);
#else
  *(uint4*)lds_dst = *(const uint4*)gsrc;
#endif
}
__device__ __forceinline__ void wait_async() {
#if HAVE_ASYNC_LDS
#if __has_builtin(__builtin_amdgcn_s_wait_asynccnt)
  __builtin_amdgcn_s_wait_asynccnt(0);
#else
  asm volatile("s_wait_asynccnt 0x0" ::: "memory");
#endif
#endif
}

// ---------------- elementwise f32 -> bf16 ----------------
__global__ __launch_bounds__(256)
void cvt_f32_bf16(const float* __restrict__ x, bf16_t* __restrict__ y, int n) {
  int i = (blockIdx.x * 256 + threadIdx.x) * 4;
  if (i + 3 < n) {
    float4 v = *(const float4*)(x + i);
    y[i + 0] = (bf16_t)v.x;
    y[i + 1] = (bf16_t)v.y;
    y[i + 2] = (bf16_t)v.z;
    y[i + 3] = (bf16_t)v.w;
  }
}

// ---------------- GEMM: C[Mr x Nc] = A[Mr x K] * BT[Nc x K]^T ----------------
// 128x128 block, 256 threads (8 waves, 4x2), K-step 32, bf16 WMMA,
// async double-buffered LDS staging.
template<int OUTF32>
__global__ __launch_bounds__(256)
void gemm_bt(const bf16_t* __restrict__ A, const bf16_t* __restrict__ BT,
             float* __restrict__ Cf, bf16_t* __restrict__ Cb,
             int Nc, int K) {
  __shared__ alignas(32) bf16_t sA[2][128][32];
  __shared__ alignas(32) bf16_t sB[2][128][32];

  const int t = threadIdx.x;
  const int l = t & 31, lane15 = l & 15, g = l >> 4;
  const int w = t >> 5, wm = w >> 1, wn = w & 1;
  const int bm = blockIdx.y * 128, bn = blockIdx.x * 128;

  const v8f vzero = {0.f,0.f,0.f,0.f,0.f,0.f,0.f,0.f};
  v8f acc[2][4];
  #pragma unroll
  for (int mt = 0; mt < 2; ++mt)
    #pragma unroll
    for (int ct = 0; ct < 4; ++ct) acc[mt][ct] = vzero;

  const int srow = t >> 1, sh = (t & 1) * 16;
  const bf16_t* Ag = A  + (size_t)(bm + srow) * K + sh;
  const bf16_t* Bg = BT + (size_t)(bn + srow) * K + sh;

  auto stage = [&](int buf, int k0) {
    async_copy16(&sA[buf][srow][sh],     Ag + k0);
    async_copy16(&sA[buf][srow][sh + 8], Ag + k0 + 8);
    async_copy16(&sB[buf][srow][sh],     Bg + k0);
    async_copy16(&sB[buf][srow][sh + 8], Bg + k0 + 8);
  };

  stage(0, 0);
  wait_async();
  __syncthreads();

  for (int k0 = 0; k0 < K; k0 += 32) {
    const int buf = (k0 >> 5) & 1;
    if (k0 + 32 < K) stage(buf ^ 1, k0 + 32);  // fill next tile while computing

    v16bf af[2], bfr[4];
    #pragma unroll
    for (int mt = 0; mt < 2; ++mt) {
      const bf16_t* ar = &sA[buf][wm * 32 + mt * 16 + lane15][0];
      af[mt] = cat8(*(const v8bf*)(ar + 8 * g), *(const v8bf*)(ar + 16 + 8 * g));
    }
    #pragma unroll
    for (int ct = 0; ct < 4; ++ct)
      bfr[ct] = cat8(*(const v8bf*)&sB[buf][wn * 64 + ct * 16 + lane15][8 * g], *(const v8bf*)&sB[buf][wn * 64 + ct * 16 + lane15][16 + 8 * g]);
    #pragma unroll
    for (int mt = 0; mt < 2; ++mt)
      #pragma unroll
      for (int ct = 0; ct < 4; ++ct)
        acc[mt][ct] = wmma_bf16(af[mt], bfr[ct], acc[mt][ct]);

    wait_async();
    __syncthreads();
  }

  #pragma unroll
  for (int mt = 0; mt < 2; ++mt)
    #pragma unroll
    for (int ct = 0; ct < 4; ++ct)
      #pragma unroll
      for (int r = 0; r < 8; ++r) {
        int row = bm + wm * 32 + mt * 16 + r + 8 * g;
        int col = bn + wn * 64 + ct * 16 + lane15;
        if (OUTF32) Cf[(size_t)row * Nc + col] = acc[mt][ct][r];
        else        Cb[(size_t)row * Nc + col] = (bf16_t)acc[mt][ct][r];
      }
}

// ---------------- head-layout gathers ----------------
// qh[b][n][l][d] = qp[b*L+l][d*16+n]
__global__ __launch_bounds__(256)
void gather_q(const bf16_t* __restrict__ qp, bf16_t* __restrict__ qhd) {
  int i = blockIdx.x * 256 + threadIdx.x;     // over B*NH*LL*HD
  int d = i & 63;
  int tmp = i >> 6;
  int lr = tmp & (LL - 1);
  int bn = tmp >> 11;
  int n = bn & 15, b = bn >> 4;
  qhd[i] = qp[((size_t)(b * LL + lr)) * DIMC + d * 16 + n];
}
// kh[b][n][m][d] and khT[b][n][d][m]
__global__ __launch_bounds__(256)
void gather_k(const bf16_t* __restrict__ kp, bf16_t* __restrict__ khd,
              bf16_t* __restrict__ khT) {
  int i = blockIdx.x * 256 + threadIdx.x;     // over B*NH*MMK*HD
  int d = i & 63;
  int tmp = i >> 6;
  int m = tmp & (MMK - 1);
  int bn = tmp >> 11;
  int n = bn & 15, b = bn >> 4;
  bf16_t vv = kp[((size_t)(b * MMK + m)) * DIMC + d * 16 + n];
  khd[i] = vv;
  khT[((size_t)bn * HD + d) * MMK + m] = vv;
}

// ---------------- flash attention (scores softmax, then P @ K) ----------------
// grid (LL/64, NH, BB), 128 threads = 4 waves; wave w owns rows [w*16, w*16+16)
// Async double-buffered K / K^T chunk staging.
__global__ __launch_bounds__(128)
void flash_attn(const bf16_t* __restrict__ qh, const bf16_t* __restrict__ kh,
                const bf16_t* __restrict__ khT, bf16_t* __restrict__ xo) {
  __shared__ alignas(32) bf16_t sK [2][64][64];   // K chunk  [m][d]
  __shared__ alignas(32) bf16_t sKT[2][64][64];   // K chunk transposed [d][m]
  __shared__ alignas(32) bf16_t sP [4][16][64];   // per-wave P strip

  const int t = threadIdx.x;
  const int l = t & 31, lane15 = l & 15, g = l >> 4;
  const int w = t >> 5;
  const int nhead = blockIdx.y, b = blockIdx.z;
  const int bh = b * NH + nhead;
  const int l0 = blockIdx.x * 64;

  const int srow = t >> 1, sh = (t & 1) * 32;
  const bf16_t* Kg  = kh  + ((size_t)bh * MMK + srow) * HD + sh;
  const bf16_t* KTg = khT + ((size_t)bh * HD  + srow) * MMK + sh;

  auto stage = [&](int buf, int m0) {
    #pragma unroll
    for (int u = 0; u < 4; ++u) {
      async_copy16(&sK [buf][srow][sh + u * 8], Kg  + (size_t)m0 * HD + u * 8);
      async_copy16(&sKT[buf][srow][sh + u * 8], KTg + m0 + u * 8);
    }
  };

  stage(0, 0);  // fill first chunk while we fetch Q fragments

  // Q fragments for this wave's 16-row strip (kept in registers)
  v16bf qf[2];
  {
    const bf16_t* Q = qh + ((size_t)bh * LL + l0 + w * 16 + lane15) * HD;
    #pragma unroll
    for (int ks = 0; ks < 2; ++ks)
      qf[ks] = cat8(*(const v8bf*)(Q + ks * 32 + 8 * g),
                    *(const v8bf*)(Q + ks * 32 + 16 + 8 * g));
  }

  const v8f vzero = {0.f,0.f,0.f,0.f,0.f,0.f,0.f,0.f};
  v8f xacc[4] = {vzero, vzero, vzero, vzero};
  float m_i[8], l_i[8];
  #pragma unroll
  for (int r = 0; r < 8; ++r) { m_i[r] = -1e30f; l_i[r] = 0.f; }

  const float Csc = 0.125f * 1.44269504089f;   // 1/sqrt(64) * log2(e)

  wait_async();
  __syncthreads();

  for (int m0 = 0; m0 < MMK; m0 += 64) {
    const int buf = (m0 >> 6) & 1;
    if (m0 + 64 < MMK) stage(buf ^ 1, m0 + 64);  // fill next chunk async

    // S strip (16 x 64) = Q strip @ K^T
    v8f sacc[4] = {vzero, vzero, vzero, vzero};
    #pragma unroll
    for (int ks = 0; ks < 2; ++ks)
      #pragma unroll
      for (int ct = 0; ct < 4; ++ct) {
        v16bf bfr = cat8(*(const v8bf*)&sK[buf][ct * 16 + lane15][ks * 32 + 8 * g], *(const v8bf*)&sK[buf][ct * 16 + lane15][ks * 32 + 16 + 8 * g]);
        sacc[ct] = wmma_bf16(qf[ks], bfr, sacc[ct]);
      }

    // online softmax: row stats via intra-16-lane butterflies
    float alpha[8];
    #pragma unroll
    for (int r = 0; r < 8; ++r) {
      float v = fmaxf(fmaxf(sacc[0][r], sacc[1][r]), fmaxf(sacc[2][r], sacc[3][r]));
      v = fmaxf(v, __shfl_xor(v, 1));
      v = fmaxf(v, __shfl_xor(v, 2));
      v = fmaxf(v, __shfl_xor(v, 4));
      v = fmaxf(v, __shfl_xor(v, 8));
      float mn = fmaxf(m_i[r], v * Csc);
      alpha[r] = exp2f(m_i[r] - mn);
      m_i[r] = mn;
    }
    float rs[8] = {0.f,0.f,0.f,0.f,0.f,0.f,0.f,0.f};
    #pragma unroll
    for (int ct = 0; ct < 4; ++ct)
      #pragma unroll
      for (int r = 0; r < 8; ++r) {
        float p = exp2f(sacc[ct][r] * Csc - m_i[r]);
        rs[r] += p;
        sP[w][r + 8 * g][ct * 16 + lane15] = (bf16_t)p;
      }
    #pragma unroll
    for (int r = 0; r < 8; ++r) {
      float s = rs[r];
      s += __shfl_xor(s, 1);
      s += __shfl_xor(s, 2);
      s += __shfl_xor(s, 4);
      s += __shfl_xor(s, 8);
      l_i[r] = l_i[r] * alpha[r] + s;
      #pragma unroll
      for (int ct = 0; ct < 4; ++ct) xacc[ct][r] *= alpha[r];
    }

    // X += P @ Kchunk   (B fragments from transposed K)
    #pragma unroll
    for (int ks = 0; ks < 2; ++ks) {
      const bf16_t* pr = &sP[w][lane15][ks * 32];
      v16bf af = cat8(*(const v8bf*)(pr + 8 * g), *(const v8bf*)(pr + 16 + 8 * g));
      #pragma unroll
      for (int ct = 0; ct < 4; ++ct) {
        v16bf bfr = cat8(*(const v8bf*)&sKT[buf][ct * 16 + lane15][ks * 32 + 8 * g], *(const v8bf*)&sKT[buf][ct * 16 + lane15][ks * 32 + 16 + 8 * g]);
        xacc[ct] = wmma_bf16(af, bfr, xacc[ct]);
      }
    }

    wait_async();
    __syncthreads();
  }

  // normalize and scatter to x[b][row][d*16 + n]
  float inv[8];
  #pragma unroll
  for (int r = 0; r < 8; ++r) inv[r] = 1.0f / l_i[r];
  #pragma unroll
  for (int ct = 0; ct < 4; ++ct)
    #pragma unroll
    for (int r = 0; r < 8; ++r) {
      int row = l0 + w * 16 + r + 8 * g;
      int d = ct * 16 + lane15;
      xo[((size_t)b * LL + row) * DIMC + d * 16 + nhead] = (bf16_t)(xacc[ct][r] * inv[r]);
    }
}

// ---------------- launcher ----------------
extern "C" void kernel_launch(void* const* d_in, const int* in_sizes, int n_in,
                              void* d_out, int out_size, void* d_ws, size_t ws_size,
                              hipStream_t stream) {
  (void)in_sizes; (void)n_in; (void)out_size; (void)ws_size;
  const float* q  = (const float*)d_in[0];
  const float* k  = (const float*)d_in[1];
  // d_in[2] = v   : dead in the reference (output never uses vp)
  const float* Wq = (const float*)d_in[3];
  const float* Wk = (const float*)d_in[4];
  // d_in[5] = Wv  : dead
  const float* Wo = (const float*)d_in[6];
  float* out = (float*)d_out;

  char* ws = (char*)d_ws;
  size_t off = 0;
  auto take = [&](size_t elems) -> bf16_t* {
    bf16_t* p = (bf16_t*)(ws + off);
    off += ((elems * sizeof(bf16_t) + 255) & ~(size_t)255);
    return p;
  };

  const size_t NQ = (size_t)BB * LL  * DIMC;  // 4M elems
  const size_t NK = (size_t)BB * MMK * DIMC;  // 4M elems
  const size_t NW = (size_t)DIMC * DIMC;      // 1M elems

  bf16_t* qb  = take(NQ);
  bf16_t* kb  = take(NK);
  bf16_t* wqb = take(NW);
  bf16_t* wkb = take(NW);
  bf16_t* wob = take(NW);
  bf16_t* qp  = take(NQ);
  bf16_t* kp  = take(NK);
  bf16_t* qhd = take(NQ);
  bf16_t* khd = take(NK);
  bf16_t* kht = take(NK);
  bf16_t* xb  = take(NQ);

  cvt_f32_bf16<<<(int)(NQ / 1024), 256, 0, stream>>>(q,  qb,  (int)NQ);
  cvt_f32_bf16<<<(int)(NK / 1024), 256, 0, stream>>>(k,  kb,  (int)NK);
  cvt_f32_bf16<<<(int)(NW / 1024), 256, 0, stream>>>(Wq, wqb, (int)NW);
  cvt_f32_bf16<<<(int)(NW / 1024), 256, 0, stream>>>(Wk, wkb, (int)NW);
  cvt_f32_bf16<<<(int)(NW / 1024), 256, 0, stream>>>(Wo, wob, (int)NW);

  dim3 gproj(DIMC / 128, (BB * LL) / 128);  // (8, 32)
  gemm_bt<0><<<gproj, 256, 0, stream>>>(qb, wqb, nullptr, qp, DIMC, DIMC);
  gemm_bt<0><<<gproj, 256, 0, stream>>>(kb, wkb, nullptr, kp, DIMC, DIMC);

  gather_q<<<(int)(NQ / 256), 256, 0, stream>>>(qp, qhd);
  gather_k<<<(int)(NK / 256), 256, 0, stream>>>(kp, khd, kht);

  flash_attn<<<dim3(LL / 64, NH, BB), 128, 0, stream>>>(qhd, khd, kht, xb);

  gemm_bt<1><<<gproj, 256, 0, stream>>>(xb, wob, out, nullptr, DIMC, DIMC);
}
